// CrossAttention_71906342469879
// MI455X (gfx1250) — matrix operand order repaired and checked by reading, not compiled, not run on hardware
//
#include <hip/hip_runtime.h>

#define B_  4
#define S_  1024
#define D_  1024
#define H_  16
#define DH_ 64

typedef __attribute__((ext_vector_type(16))) _Float16 v16h;
typedef __attribute__((ext_vector_type(8)))  _Float16 h8;
typedef __attribute__((ext_vector_type(4)))  _Float16 h4;
typedef __attribute__((ext_vector_type(8)))  float    v8f;
typedef __attribute__((ext_vector_type(4)))  float    f4;

static __device__ __forceinline__ v16h cat16(h8 lo, h8 hi) {
  return __builtin_shufflevector(lo, hi, 0,1,2,3,4,5,6,7,8,9,10,11,12,13,14,15);
}

static __device__ __forceinline__ float redmax16(float v) {
  v = fmaxf(v, __shfl_xor(v, 1, 16));
  v = fmaxf(v, __shfl_xor(v, 2, 16));
  v = fmaxf(v, __shfl_xor(v, 4, 16));
  v = fmaxf(v, __shfl_xor(v, 8, 16));
  return v;
}
static __device__ __forceinline__ float redsum16(float v) {
  v += __shfl_xor(v, 1, 16);
  v += __shfl_xor(v, 2, 16);
  v += __shfl_xor(v, 4, 16);
  v += __shfl_xor(v, 8, 16);
  return v;
}

// LDS byte address of a generic pointer to __shared__ (aperture: addr[31:0])
static __device__ __forceinline__ unsigned lds_addr(const void* p) {
  return (unsigned)(unsigned long long)p;
}

// Async global -> LDS copy, 16B per lane, GVS mode; tracked by ASYNCcnt.
static __device__ __forceinline__ void async_copy_b128(unsigned lds,
                                                       unsigned goff,
                                                       unsigned long long gbase) {
  asm volatile("global_load_async_to_lds_b128 %0, %1, %2"
               :: "v"(lds), "v"(goff), "s"(gbase) : "memory");
}
static __device__ __forceinline__ void wait_async0() {
  asm volatile("s_wait_asynccnt 0x0" ::: "memory");
}

// ---------------------------------------------------------------------------
// Kernel 1: Out[b,h,s,dh] (f16) = X[b,s,:] @ W[n,:]^T + bias[n]
// 128x128 tile / 256-thread WG; 8 waves each 32(M) x 64(N).
// Double-buffered LDS: global f32 loads for step i+1 overlap WMMAs of step i.
// ---------------------------------------------------------------------------
__global__ __launch_bounds__(256)
void xattn_proj_kernel(const float* __restrict__ X,
                       const float* __restrict__ W,
                       const float* __restrict__ bias,
                       _Float16* __restrict__ Out) {
  constexpr int SA = 40;                       // padded stride (halves) = 80B
  __shared__ __align__(16) _Float16 As[2][128 * SA];
  __shared__ __align__(16) _Float16 Bs[2][128 * SA];

  const int tid  = threadIdx.x;
  const int w    = tid >> 5;
  const int lane = tid & 31;
  const int lr   = lane & 15;
  const int half = lane >> 4;
  const int m0   = blockIdx.y * 128;
  const int n0   = blockIdx.x * 128;
  const int wm   = (w >> 1) * 32;
  const int wn   = (w & 1) * 64;

  v8f acc[2][4];
  const v8f zero8 = {0.f,0.f,0.f,0.f,0.f,0.f,0.f,0.f};
#pragma unroll
  for (int ms = 0; ms < 2; ++ms)
#pragma unroll
    for (int ns = 0; ns < 4; ++ns) acc[ms][ns] = zero8;

  f4 xr[4], wr[4];
  const int srow = tid >> 1;                   // staging row (2 thr / row)
  const int skq  = (tid & 1) * 16;             // not used; see chunk mapping

  // chunk mapping: thread t, rep j -> float4 chunk i = t + 256*j
  auto gload = [&](int k0) {
#pragma unroll
    for (int j = 0; j < 4; ++j) {
      int i = tid + 256 * j, row = i >> 3, kq = (i & 7) * 4;
      xr[j] = *(const f4*)(X + (size_t)(m0 + row) * D_ + k0 + kq);
      wr[j] = *(const f4*)(W + (size_t)(n0 + row) * D_ + k0 + kq);
    }
  };
  auto lstore = [&](int p) {
#pragma unroll
    for (int j = 0; j < 4; ++j) {
      int i = tid + 256 * j, row = i >> 3, kq = (i & 7) * 4;
      h4 hx; hx[0]=(_Float16)xr[j][0]; hx[1]=(_Float16)xr[j][1];
             hx[2]=(_Float16)xr[j][2]; hx[3]=(_Float16)xr[j][3];
      *(h4*)(&As[p][row * SA + kq]) = hx;
      h4 hw; hw[0]=(_Float16)wr[j][0]; hw[1]=(_Float16)wr[j][1];
             hw[2]=(_Float16)wr[j][2]; hw[3]=(_Float16)wr[j][3];
      *(h4*)(&Bs[p][row * SA + kq]) = hw;
    }
  };

  (void)srow; (void)skq;
  gload(0);
  lstore(0);
  __syncthreads();

  for (int it = 0; it < D_ / 32; ++it) {
    const int p = it & 1;
    if (it + 1 < D_ / 32) gload((it + 1) * 32);   // prefetch next k-step

    v16h a[2], b[4];
#pragma unroll
    for (int ms = 0; ms < 2; ++ms) {
      int row = wm + ms * 16 + lr;
      int kb  = half * 8;
      h8 x0 = *(const h8*)(&As[p][row * SA + kb]);
      h8 x1 = *(const h8*)(&As[p][row * SA + kb + 16]);
      a[ms] = cat16(x0, x1);
    }
#pragma unroll
    for (int ns = 0; ns < 4; ++ns) {
      int row = wn + ns * 16 + lr;
      int kb  = half * 8;
      h8 x0 = *(const h8*)(&Bs[p][row * SA + kb]);
      h8 x1 = *(const h8*)(&Bs[p][row * SA + kb + 16]);
      b[ns] = cat16(x0, x1);
    }
#pragma unroll
    for (int ms = 0; ms < 2; ++ms)
#pragma unroll
      for (int ns = 0; ns < 4; ++ns)
        acc[ms][ns] = __builtin_amdgcn_wmma_f32_16x16x32_f16(
            false, a[ms], false, b[ns], (short)0, acc[ms][ns], false, false);

    if (it + 1 < D_ / 32) lstore(1 - p);          // fill other buffer
    __syncthreads();
  }

  // epilogue: f16 head layout [B,H,S,DH]
#pragma unroll
  for (int ms = 0; ms < 2; ++ms)
#pragma unroll
    for (int ns = 0; ns < 4; ++ns) {
      int ncol = n0 + wn + ns * 16 + lr;
      float bv = bias[ncol];
      int hh = ncol >> 6, dh = ncol & 63;
#pragma unroll
      for (int r = 0; r < 8; ++r) {
        int m  = m0 + wm + ms * 16 + r + half * 8;
        int bb = m >> 10, s = m & 1023;
        float v = acc[ms][ns][r] + bv;
        Out[(((size_t)bb * H_ + hh) * S_ + s) * DH_ + dh] = (_Float16)v;
      }
    }
}

// ---------------------------------------------------------------------------
// Kernel 2: flash attention, 128 q-rows / WG (16 per wave), 64-key tiles.
// K tile staged via GLOBAL_LOAD_ASYNC_TO_LDS_B128 (ASYNCcnt), V tile staged
// transposed through VGPRs; both double-buffered so tile t+1 transfers
// overlap tile t WMMAs. Online softmax in the f32 accumulator layout.
// ---------------------------------------------------------------------------
__global__ __launch_bounds__(256)
void xattn_flash_kernel(const _Float16* __restrict__ Qh,
                        const _Float16* __restrict__ Kh,
                        const _Float16* __restrict__ Vh,
                        const int* __restrict__ mask,
                        float* __restrict__ Out) {
  constexpr int SK = 72;                       // padded stride (halves) = 144B
  __shared__ __align__(16) _Float16 Ks[2][64 * SK];     // K tile [key][dh]
  __shared__ __align__(16) _Float16 Vt[2][64 * SK];     // V^T    [dh][key]
  __shared__ __align__(16) _Float16 Pw[8 * 16 * SK];    // per-wave P scratch

  const int tid  = threadIdx.x;
  const int w    = tid >> 5;
  const int lane = tid & 31;
  const int lr   = lane & 15;
  const int half = lane >> 4;
  const int bh   = blockIdx.y;
  const int b    = bh >> 4;
  const int h    = bh & 15;
  const int q0   = blockIdx.x * 128;
  const int qrb  = q0 + w * 16;

  const _Float16* Qb = Qh + (size_t)bh * S_ * DH_;
  const _Float16* Kb = Kh + (size_t)bh * S_ * DH_;
  const _Float16* Vb = Vh + (size_t)bh * S_ * DH_;
  const unsigned long long kb64 = (unsigned long long)Kb;

  // Q fragments resident in registers for the whole kernel
  v16h aq[2];
#pragma unroll
  for (int t = 0; t < 2; ++t) {
    int kb = t * 32 + half * 8;
    const _Float16* p = Qb + (size_t)(qrb + lr) * DH_ + kb;
    aq[t] = cat16(*(const h8*)p, *(const h8*)(p + 16));
  }

  const v8f zero8 = {0.f,0.f,0.f,0.f,0.f,0.f,0.f,0.f};
  v8f acco[4];
#pragma unroll
  for (int ds = 0; ds < 4; ++ds) acco[ds] = zero8;
  float mrow[8], lrow[8];
#pragma unroll
  for (int r = 0; r < 8; ++r) { mrow[r] = -3.0e38f; lrow[r] = 0.f; }

  _Float16* Pme = &Pw[w * 16 * SK];
  const size_t mbase = ((size_t)b * S_ + qrb) * S_;

  const int srow = tid >> 3;                   // staging: i = tid + 256*j
  h8 vr[2];

  // ---- prologue: stage tile 0 into buffer 0 ----
#pragma unroll
  for (int j = 0; j < 2; ++j) {
    int i = tid + 256 * j, row = i >> 3, c = i & 7;
    async_copy_b128(lds_addr(&Ks[0][row * SK + c * 8]),
                    (unsigned)((row * DH_ + c * 8) * 2), kb64);
    vr[j] = *(const h8*)(Vb + (size_t)row * DH_ + c * 8);
  }
#pragma unroll
  for (int j = 0; j < 2; ++j) {
    int i = tid + 256 * j, row = i >> 3, c = i & 7;
#pragma unroll
    for (int e = 0; e < 8; ++e) Vt[0][(c * 8 + e) * SK + row] = vr[j][e];
  }
  wait_async0();
  __syncthreads();
  (void)srow;

  for (int kt = 0; kt < S_ / 64; ++kt) {
    const int p  = kt & 1;
    const int k0 = kt * 64;

    // ---- issue transfers for tile kt+1 into buffer 1-p (no wait yet) ----
    if (kt + 1 < S_ / 64) {
      const int k1 = (kt + 1) * 64;
#pragma unroll
      for (int j = 0; j < 2; ++j) {
        int i = tid + 256 * j, row = i >> 3, c = i & 7;
        async_copy_b128(lds_addr(&Ks[1 - p][row * SK + c * 8]),
                        (unsigned)(((k1 + row) * DH_ + c * 8) * 2), kb64);
        vr[j] = *(const h8*)(Vb + (size_t)(k1 + row) * DH_ + c * 8);
      }
    }

    // ---- preload mask tile so the latency hides under the WMMAs ----
    int mv[4][8];
#pragma unroll
    for (int ns = 0; ns < 4; ++ns) {
      int kc = k0 + ns * 16 + lr;
#pragma unroll
      for (int r = 0; r < 8; ++r)
        mv[ns][r] = mask[mbase + (size_t)(r + half * 8) * S_ + kc];
    }

    // ---- scores: S = Q @ K^T (16 x 64), 8 WMMAs ----
    v8f sc[4];
#pragma unroll
    for (int ns = 0; ns < 4; ++ns) sc[ns] = zero8;
#pragma unroll
    for (int t = 0; t < 2; ++t)
#pragma unroll
      for (int ns = 0; ns < 4; ++ns) {
        const _Float16* kp = &Ks[p][(ns * 16 + lr) * SK + t * 32 + half * 8];
        v16h bf = cat16(*(const h8*)kp, *(const h8*)(kp + 16));
        sc[ns] = __builtin_amdgcn_wmma_f32_16x16x32_f16(
            false, aq[t], false, bf, (short)0, sc[ns], false, false);
      }

    // ---- scale + mask ----
#pragma unroll
    for (int ns = 0; ns < 4; ++ns)
#pragma unroll
      for (int r = 0; r < 8; ++r) {
        float v = sc[ns][r] * 0.125f;          // 1/sqrt(DH)
        sc[ns][r] = (mv[ns][r] == 0) ? -1.0e9f : v;
      }

    // ---- online softmax (per row: vgpr r within the 16-lane half) ----
    float corr[8], rsum[8];
#pragma unroll
    for (int r = 0; r < 8; ++r) {
      float v = fmaxf(fmaxf(sc[0][r], sc[1][r]), fmaxf(sc[2][r], sc[3][r]));
      v = redmax16(v);
      float mnew = fmaxf(mrow[r], v);
      corr[r] = __expf(mrow[r] - mnew);
      mrow[r] = mnew;
      rsum[r] = 0.f;
    }
#pragma unroll
    for (int ds = 0; ds < 4; ++ds)
#pragma unroll
      for (int r = 0; r < 8; ++r) acco[ds][r] *= corr[r];
#pragma unroll
    for (int ns = 0; ns < 4; ++ns)
#pragma unroll
      for (int r = 0; r < 8; ++r) {
        float pe = __expf(sc[ns][r] - mrow[r]);
        sc[ns][r] = pe;
        rsum[r] += pe;
      }
#pragma unroll
    for (int r = 0; r < 8; ++r)
      lrow[r] = lrow[r] * corr[r] + redsum16(rsum[r]);

    // ---- P (acc layout) -> f16 A-fragment layout via per-wave LDS ----
#pragma unroll
    for (int ns = 0; ns < 4; ++ns)
#pragma unroll
      for (int r = 0; r < 8; ++r)
        Pme[(r + half * 8) * SK + ns * 16 + lr] = (_Float16)sc[ns][r];
    asm volatile("s_wait_dscnt 0x0" ::: "memory");

    // ---- O += P @ V (16 x 64), 8 WMMAs ----
#pragma unroll
    for (int t = 0; t < 2; ++t) {
      int kb = t * 32 + half * 8;
      const _Float16* pp = &Pme[lr * SK + kb];
      v16h ap = cat16(*(const h8*)pp, *(const h8*)(pp + 16));
#pragma unroll
      for (int ds = 0; ds < 4; ++ds) {
        const _Float16* pv = &Vt[p][(ds * 16 + lr) * SK + t * 32 + half * 8];
        v16h bv = cat16(*(const h8*)pv, *(const h8*)(pv + 16));
        acco[ds] = __builtin_amdgcn_wmma_f32_16x16x32_f16(
            false, ap, false, bv, (short)0, acco[ds], false, false);
      }
    }

    // ---- scatter prefetched V into V^T buffer 1-p, fence, barrier ----
    if (kt + 1 < S_ / 64) {
#pragma unroll
      for (int j = 0; j < 2; ++j) {
        int i = tid + 256 * j, row = i >> 3, c = i & 7;
#pragma unroll
        for (int e = 0; e < 8; ++e)
          Vt[1 - p][(c * 8 + e) * SK + row] = vr[j][e];
      }
    }
    wait_async0();
    __syncthreads();
  }

  // ---- epilogue: normalize, write f32 [B,S,D] ----
#pragma unroll
  for (int ds = 0; ds < 4; ++ds) {
    int dh = ds * 16 + lr;
#pragma unroll
    for (int r = 0; r < 8; ++r) {
      int q = qrb + r + half * 8;
      Out[((size_t)b * S_ + q) * D_ + h * DH_ + dh] = acco[ds][r] / lrow[r];
    }
  }
}

extern "C" void kernel_launch(void* const* d_in, const int* in_sizes, int n_in,
                              void* d_out, int out_size, void* d_ws, size_t ws_size,
                              hipStream_t stream) {
  (void)in_sizes; (void)n_in; (void)out_size; (void)ws_size;
  const float* q    = (const float*)d_in[0];
  const float* k    = (const float*)d_in[1];
  const float* v    = (const float*)d_in[2];
  const int*   mask = (const int*)d_in[3];
  const float* Wq   = (const float*)d_in[4];
  const float* bq   = (const float*)d_in[5];
  const float* Wk   = (const float*)d_in[6];
  const float* bk   = (const float*)d_in[7];
  const float* Wv   = (const float*)d_in[8];
  const float* bv   = (const float*)d_in[9];

  const size_t HSZ = (size_t)B_ * H_ * S_ * DH_;   // 4M halves = 8MB each
  _Float16* Qh = (_Float16*)d_ws;
  _Float16* Kh = Qh + HSZ;
  _Float16* Vh = Kh + HSZ;

  dim3 blk(256);
  dim3 gp(D_ / 128, (B_ * S_) / 128);
  xattn_proj_kernel<<<gp, blk, 0, stream>>>(q, Wq, bq, Qh);
  xattn_proj_kernel<<<gp, blk, 0, stream>>>(k, Wk, bk, Kh);
  xattn_proj_kernel<<<gp, blk, 0, stream>>>(v, Wv, bv, Vh);

  dim3 ga(S_ / 128, B_ * H_);
  xattn_flash_kernel<<<ga, blk, 0, stream>>>(Qh, Kh, Vh, mask, (float*)d_out);
}
